// KG_Attention_14465449853551
// MI455X (gfx1250) — matrix operand order repaired and checked by reading, not compiled, not run on hardware
//
#include <hip/hip_runtime.h>
#include <hip/hip_bf16.h>

// ---------------------------------------------------------------------------
// CDNA5 (gfx1250) KG-attention. All matmuls via v_wmma_f32_16x16x32_f16
// (f16 data, f32 accumulate). f32->f16 once up front; GEMM tiles staged with
// double-buffered global_load_async_to_lds_b128 (+ s_wait_asynccnt pipeline).
// Fragment loads are contiguous vector loads, except GEMM W frags (per-element LDS reads).
// ---------------------------------------------------------------------------

typedef _Float16 half_t;
typedef half_t v16h __attribute__((ext_vector_type(16)));
typedef half_t v8h  __attribute__((ext_vector_type(8)));
typedef float  v8f  __attribute__((ext_vector_type(8)));
typedef int    v4i_t __attribute__((vector_size(4 * sizeof(int))));
typedef __attribute__((address_space(3))) v4i_t* lds_v4i_p;

union F16Frag { v16h v; v8h h[2]; half_t e[16]; };

#if defined(__gfx1250__) && __has_builtin(__builtin_amdgcn_global_load_async_to_lds_b128)
#define HAVE_ASYNC_LDS 1
#endif

template <int N>
__device__ __forceinline__ void wait_async() {
#if __has_builtin(__builtin_amdgcn_s_wait_asynccnt)
  __builtin_amdgcn_s_wait_asynccnt(N);
#else
  asm volatile("s_wait_asynccnt %0" ::"i"(N) : "memory");
#endif
}

// B-frag k order used here (contiguous, same as A): lanes 0-15 K=0..15, lanes 16-31 K=16..31
__device__ __forceinline__ int b_kpair(int r, int hf) {
  return hf * 16 + 2 * r;
}

// 16-lane-group butterfly via ds_swizzle (group-of-32 mode: xor<<10 | and=0x1F)
template <int XM>
__device__ __forceinline__ float swzx(float v) {
  return __int_as_float(
      __builtin_amdgcn_ds_swizzle(__float_as_int(v), (XM << 10) | 0x1F));
}
__device__ __forceinline__ float rowmax16(float v) {
  v = fmaxf(v, swzx<1>(v)); v = fmaxf(v, swzx<2>(v));
  v = fmaxf(v, swzx<4>(v)); v = fmaxf(v, swzx<8>(v));
  return v;
}
__device__ __forceinline__ float rowsum16(float v) {
  v += swzx<1>(v); v += swzx<2>(v); v += swzx<4>(v); v += swzx<8>(v);
  return v;
}

// ---------------------------------------------------------------------------
// f32 -> f16 elementwise (sizes are multiples of 1024)
// ---------------------------------------------------------------------------
__global__ __launch_bounds__(256)
void cvt_f32_f16(const float* __restrict__ in, half_t* __restrict__ out, int n4) {
  int i = blockIdx.x * 256 + threadIdx.x;
  if (i < n4) {
    float4 v = ((const float4*)in)[i];
    half_t* o = out + (size_t)i * 4;
    o[0] = (half_t)v.x; o[1] = (half_t)v.y; o[2] = (half_t)v.z; o[3] = (half_t)v.w;
  }
}

// ---------------------------------------------------------------------------
// GEMM: C[b] = A[b] (MxK f16) @ W (KxN f16) + bias(f32); OT = f16 or f32.
// 256 threads (8 waves, 2x4), tile 128x128, k-step 32, double-buffered async.
// ---------------------------------------------------------------------------
#define BM 128
#define BN 128
#define BK 32
#define APAD 8   // A row stride 40 f16 = 80B: 16B aligned, bank stride 20

template <typename OT>
__global__ __launch_bounds__(256)
void gemm_bias_f16wmma(const half_t* __restrict__ A, int lda, long long abs_,
                       const half_t* __restrict__ W, int ldw,
                       const float* __restrict__ bias,
                       OT* __restrict__ C, int ldc, long long cbs_, int K) {
  __shared__ half_t As[2][BM][BK + APAD];   // [m][k]
  __shared__ half_t Ws[2][BK][BN];          // [k][n]

  const int tid  = threadIdx.x;
  const int lane = tid & 31;
  const int ln   = lane & 15;
  const int hf   = lane >> 4;
  const int wave = tid >> 5;
  const int wr   = wave >> 2;               // 0..1 -> 64 rows
  const int wc   = wave & 3;                // 0..3 -> 32 cols
  const int row0 = blockIdx.y * BM;
  const int col0 = blockIdx.x * BN;

  const half_t* Ab = A + (long long)blockIdx.z * abs_;
  OT*           Cb = C + (long long)blockIdx.z * cbs_;

  // per-thread staging coordinates (16B chunks)
  const int ar = tid >> 2, ac = (tid & 3) * 8;    // A: 64 rows / call, 2 calls
  const int br = tid >> 4, bc = (tid & 15) * 8;   // W: 16 rows / call, 2 calls

  // stage one 128x32 A tile + 32x128 W tile into buffer `buf`
  auto stage = [&](int k0, int buf) {
#ifdef HAVE_ASYNC_LDS
    __builtin_amdgcn_global_load_async_to_lds_b128(
        (v4i_t*)(Ab + (row0 + ar) * lda + k0 + ac),
        (lds_v4i_p)&As[buf][ar][ac], 0, 0);
    __builtin_amdgcn_global_load_async_to_lds_b128(
        (v4i_t*)(Ab + (row0 + 64 + ar) * lda + k0 + ac),
        (lds_v4i_p)&As[buf][64 + ar][ac], 0, 0);
    __builtin_amdgcn_global_load_async_to_lds_b128(
        (v4i_t*)(W + (k0 + br) * ldw + col0 + bc),
        (lds_v4i_p)&Ws[buf][br][bc], 0, 0);
    __builtin_amdgcn_global_load_async_to_lds_b128(
        (v4i_t*)(W + (k0 + 16 + br) * ldw + col0 + bc),
        (lds_v4i_p)&Ws[buf][16 + br][bc], 0, 0);
#else
    *(float4*)&As[buf][ar][ac]      = *(const float4*)(Ab + (row0 + ar) * lda + k0 + ac);
    *(float4*)&As[buf][64 + ar][ac] = *(const float4*)(Ab + (row0 + 64 + ar) * lda + k0 + ac);
    *(float4*)&Ws[buf][br][bc]      = *(const float4*)(W + (k0 + br) * ldw + col0 + bc);
    *(float4*)&Ws[buf][16 + br][bc] = *(const float4*)(W + (k0 + 16 + br) * ldw + col0 + bc);
#endif
  };

  v8f acc[4][2] = {};
  const int T = K / BK;

  stage(0, 0);
  for (int t = 0; t < T; ++t) {
    const int buf = t & 1;
    if (t + 1 < T) {
      stage((t + 1) * BK, buf ^ 1);
#ifdef HAVE_ASYNC_LDS
      wait_async<4>();      // first 4 (tile t) complete; tile t+1 in flight
#endif
    } else {
#ifdef HAVE_ASYNC_LDS
      wait_async<0>();
#endif
    }
    __syncthreads();

    // A-frag = 16 consecutive f16 per lane (two 16B LDS chunks)
    F16Frag afrag[4], bfrag[2];
#pragma unroll
    for (int mi = 0; mi < 4; ++mi) {
      int m = wr * 64 + mi * 16 + ln;
      afrag[mi].h[0] = *(const v8h*)&As[buf][m][hf * 16];
      afrag[mi].h[1] = *(const v8h*)&As[buf][m][hf * 16 + 8];
    }
#pragma unroll
    for (int ni = 0; ni < 2; ++ni) {
      int n = wc * 32 + ni * 16 + ln;
#pragma unroll
      for (int r = 0; r < 8; ++r) {
        int kb = b_kpair(r, hf);
        bfrag[ni].e[2 * r]     = Ws[buf][kb][n];
        bfrag[ni].e[2 * r + 1] = Ws[buf][kb + 1][n];
      }
    }
#pragma unroll
    for (int mi = 0; mi < 4; ++mi)
#pragma unroll
      for (int ni = 0; ni < 2; ++ni)
        acc[mi][ni] = __builtin_amdgcn_wmma_f32_16x16x32_f16(
            false, afrag[mi].v, false, bfrag[ni].v, (short)0, acc[mi][ni],
            false, false);
    __syncthreads();
  }

  // C/D frag: col = lane%16, row = reg + 8*(lane/16)
#pragma unroll
  for (int mi = 0; mi < 4; ++mi)
#pragma unroll
    for (int ni = 0; ni < 2; ++ni)
#pragma unroll
      for (int r = 0; r < 8; ++r) {
        int m = row0 + wr * 64 + mi * 16 + r + 8 * hf;
        int n = col0 + wc * 32 + ni * 16 + ln;
        Cb[(long long)m * ldc + n] = (OT)(acc[mi][ni][r] + bias[n]);
      }
}

// ---------------------------------------------------------------------------
// Flash-style attention, f16 in / f16 out, 1 wave per 16 query rows of (b,h).
// 32-key tiles: 4 WMMAs for scores (K-dim 64), 4 WMMAs for P@V (d = 4x16).
// V tile staged transposed in LDS so all fragments are contiguous b128 loads.
// Causal masking reproduces  w*b - 10000*(1-b)  exactly.
// ---------------------------------------------------------------------------
__global__ __launch_bounds__(32)
void attn_f16wmma(const half_t* __restrict__ Qbuf, int qld, long long qbs,
                  const half_t* __restrict__ KVbuf, int kvld, long long kvbs,
                  int kcol, int vcol, int nkeys, int causal,
                  half_t* __restrict__ Obuf, int oldc, long long obs) {
  __shared__ half_t Pl[16][40];   // P tile, row stride 80B (16B aligned chunks)
  __shared__ half_t Vt[64][40];   // V^T tile: [d][key-local]

  const int lane = threadIdx.x;
  const int ln   = lane & 15;
  const int hf   = lane >> 4;
  const int q0   = blockIdx.x * 16;
  const int h    = blockIdx.y;
  const int b    = blockIdx.z;

  const half_t* Q  = Qbuf  + (long long)b * qbs  + h * 64;
  const half_t* Kp = KVbuf + (long long)b * kvbs + kcol + h * 64;
  const half_t* Vp = KVbuf + (long long)b * kvbs + vcol + h * 64;
  half_t*       O  = Obuf  + (long long)b * obs  + h * 64;

  // Q tile [16x64] as two A-fragments: contiguous 16B chunks per lane
  F16Frag aq[2];
#pragma unroll
  for (int c = 0; c < 2; ++c) {
    const half_t* qp = Q + (q0 + ln) * qld + c * 32;
    aq[c].h[0] = *(const v8h*)(qp + hf * 16);
    aq[c].h[1] = *(const v8h*)(qp + hf * 16 + 8);
  }

  float mrun[8], lrun[8];
#pragma unroll
  for (int r = 0; r < 8; ++r) { mrun[r] = -3.0e38f; lrun[r] = 0.0f; }
  v8f accO[4] = {};

  const int   kmax  = causal ? (q0 + 16) : nkeys;
  const float scale = 0.125f;   // 1/sqrt(64)

  for (int kt = 0; kt < kmax; kt += 32) {
    // ---- stage V tile transposed: lane owns key row (kt+lane), 8x b128 ----
#pragma unroll
    for (int c = 0; c < 8; ++c) {
      v8h t = *(const v8h*)(Vp + (kt + lane) * kvld + c * 8);
#pragma unroll
      for (int j = 0; j < 8; ++j) Vt[c * 8 + j][lane] = t[j];
    }

    // ---- scores: K B-frag = 32 contiguous bytes per lane ------------------
    v8f s[2] = {};
#pragma unroll
    for (int f = 0; f < 2; ++f)
#pragma unroll
      for (int c = 0; c < 2; ++c) {
        F16Frag bk;
        bk.v = *(const v16h*)(Kp + (kt + 16 * f + ln) * kvld + c * 32 + hf * 16);
        s[f] = __builtin_amdgcn_wmma_f32_16x16x32_f16(
            false, aq[c].v, false, bk.v, (short)0, s[f], false, false);
      }

    // ---- scale + causal mask + online-softmax stats -----------------------
    float tmax[8];
#pragma unroll
    for (int r = 0; r < 8; ++r) {
      int row = q0 + r + 8 * hf;
      float v0 = s[0][r] * scale;
      float v1 = s[1][r] * scale;
      if (causal) {
        if (kt + ln      > row) v0 = -10000.0f;
        if (kt + 16 + ln > row) v1 = -10000.0f;
      }
      s[0][r] = v0; s[1][r] = v1;
      tmax[r] = rowmax16(fmaxf(v0, v1));
    }
#pragma unroll
    for (int r = 0; r < 8; ++r) {
      float nm    = fmaxf(mrun[r], tmax[r]);
      float alpha = __expf(mrun[r] - nm);
      mrun[r] = nm;
      float p0 = __expf(s[0][r] - nm);
      float p1 = __expf(s[1][r] - nm);
      lrun[r] = lrun[r] * alpha + rowsum16(p0 + p1);
#pragma unroll
      for (int d = 0; d < 4; ++d) accO[d][r] *= alpha;
      int m = r + 8 * hf;
      Pl[m][ln]      = (half_t)p0;     // C-frag -> row-major relayout
      Pl[m][16 + ln] = (half_t)p1;
    }
    __syncthreads();

    // ---- P A-frag + 4 V B-frags: all contiguous b128 LDS reads ------------
    F16Frag pa;
    pa.h[0] = *(const v8h*)&Pl[ln][hf * 16];
    pa.h[1] = *(const v8h*)&Pl[ln][hf * 16 + 8];
    F16Frag bv[4];
#pragma unroll
    for (int db = 0; db < 4; ++db) {
      bv[db].h[0] = *(const v8h*)&Vt[db * 16 + ln][hf * 16];
      bv[db].h[1] = *(const v8h*)&Vt[db * 16 + ln][hf * 16 + 8];
    }
    __syncthreads();

#pragma unroll
    for (int db = 0; db < 4; ++db)
      accO[db] = __builtin_amdgcn_wmma_f32_16x16x32_f16(
          false, pa.v, false, bv[db].v, (short)0, accO[db], false, false);
  }

#pragma unroll
  for (int db = 0; db < 4; ++db)
#pragma unroll
    for (int r = 0; r < 8; ++r) {
      int m = q0 + r + 8 * hf;
      O[(m * oldc) + db * 16 + ln] = (half_t)(accO[db][r] / lrun[r]);
    }
}

// ---------------------------------------------------------------------------
// Host orchestration
// ---------------------------------------------------------------------------
extern "C" void kernel_launch(void* const* d_in, const int* in_sizes, int n_in,
                              void* d_out, int out_size, void* d_ws, size_t ws_size,
                              hipStream_t stream) {
  (void)in_sizes; (void)n_in; (void)out_size; (void)ws_size;

  const int B = 4, S = 1024, D = 1024, H = 16, SKG = 256;

  const float* x       = (const float*)d_in[0];
  const float* pred_kg = (const float*)d_in[1];
  const float* Wc      = (const float*)d_in[2];
  const float* bc      = (const float*)d_in[3];
  const float* Wk      = (const float*)d_in[4];
  const float* bk      = (const float*)d_in[5];
  const float* Wp      = (const float*)d_in[6];
  const float* bp      = (const float*)d_in[7];
  float* out = (float*)d_out;

  // f16 workspace layout
  half_t* xh   = (half_t*)d_ws;                          // B*S*D
  half_t* pkh  = xh   + (size_t)B * S * D;               // B*SKG*D
  half_t* Wch  = pkh  + (size_t)B * SKG * D;             // D*3D
  half_t* Wkh  = Wch  + (size_t)D * 3 * D;               // D*2D
  half_t* Wph  = Wkh  + (size_t)D * 2 * D;               // D*D
  half_t* qkvh = Wph  + (size_t)D * D;                   // B*S*3D
  half_t* kkvh = qkvh + (size_t)B * S * 3 * D;           // B*SKG*2D
  half_t* ah   = kkvh + (size_t)B * SKG * 2 * D;         // B*S*D
  half_t* kah  = ah   + (size_t)B * S * D;               // B*S*D

  // 0) one-shot f32 -> f16 conversions (bandwidth-trivial at 23.3 TB/s)
  auto cvt = [&](const float* src, half_t* dst, long long n) {
    int n4 = (int)(n / 4);
    cvt_f32_f16<<<dim3((n4 + 255) / 256), 256, 0, stream>>>(src, dst, n4);
  };
  cvt(x,       xh,  (long long)B * S * D);
  cvt(pred_kg, pkh, (long long)B * SKG * D);
  cvt(Wc,      Wch, (long long)D * 3 * D);
  cvt(Wk,      Wkh, (long long)D * 2 * D);
  cvt(Wp,      Wph, (long long)D * D);

  // 1) qkv = x @ Wc + bc  (M=4096, N=3072, K=1024) -> f16
  gemm_bias_f16wmma<half_t><<<dim3(3 * D / BN, B * S / BM, 1), 256, 0, stream>>>(
      xh, D, 0LL, Wch, 3 * D, bc, qkvh, 3 * D, 0LL, D);

  // 2) kkv = pred_kg @ Wk + bk  (M=1024, N=2048, K=1024) -> f16
  gemm_bias_f16wmma<half_t><<<dim3(2 * D / BN, B * SKG / BM, 1), 256, 0, stream>>>(
      pkh, D, 0LL, Wkh, 2 * D, bk, kkvh, 2 * D, 0LL, D);

  // 3) causal self-attention (K at col D, V at col 2D inside qkv)
  attn_f16wmma<<<dim3(S / 16, H, B), 32, 0, stream>>>(
      qkvh, 3 * D, (long long)S * 3 * D,
      qkvh, 3 * D, (long long)S * 3 * D,
      D, 2 * D, S, 1,
      ah, D, (long long)S * D);

  // 4) KG cross-attention (K at col 0, V at col D inside kkv), no mask
  attn_f16wmma<<<dim3(S / 16, H, B), 32, 0, stream>>>(
      qkvh, 3 * D, (long long)S * 3 * D,
      kkvh, 2 * D, (long long)SKG * 2 * D,
      0, D, SKG, 0,
      kah, D, (long long)S * D);

  // 5) out[:, 0:S, :] = a @ Wp + bp   (batched, out batch stride 2S*D, f32)
  gemm_bias_f16wmma<float><<<dim3(D / BN, S / BM, B), 256, 0, stream>>>(
      ah, D, (long long)S * D, Wph, D, bp,
      out, D, (long long)2 * S * D, D);

  // 6) out[:, S:2S, :] = ka @ Wp + bp
  gemm_bias_f16wmma<float><<<dim3(D / BN, S / BM, B), 256, 0, stream>>>(
      kah, D, (long long)S * D, Wph, D, bp,
      out + (long long)S * D, D, (long long)2 * S * D, D);
}
